// CrossAttention_7421703487990
// MI455X (gfx1250) — matrix operand order repaired and checked by reading, not compiled, not run on hardware
//
#include <hip/hip_runtime.h>
#include <hip/hip_bf16.h>

#define DEV __device__ __forceinline__

typedef __attribute__((ext_vector_type(16))) __bf16        v16bf;
typedef __attribute__((ext_vector_type(8)))  float         v8f;
typedef __attribute__((ext_vector_type(4)))  unsigned int  u32x4;
typedef __attribute__((ext_vector_type(4)))  int           i32x4;
typedef __attribute__((ext_vector_type(8)))  int           i32x8;

union FragBF16 { u32x4 q[2]; unsigned u[8]; v16bf v; };
union HalfView { u32x4 q; unsigned short h[8]; };

DEV unsigned short f2bf(float f) {
  union { float f; unsigned u; } x; x.f = f;
  unsigned r = x.u + 0x7FFFu + ((x.u >> 16) & 1u);   // RNE
  return (unsigned short)(r >> 16);
}

DEV v8f wmma_bf16(v16bf a, v16bf b, v8f c) {
  return __builtin_amdgcn_wmma_f32_16x16x32_bf16(false, a, false, b, (short)0, c,
                                                 false, false);
}

// Async global->LDS copy of 16B per lane (ASYNCcnt path). INST_OFFSET is added
// to both the LDS and the global address per the ISA pseudocode.
DEV void async_copy_b128(unsigned lds_byte_addr, const void* gaddr) {
  asm volatile("global_load_async_to_lds_b128 %0, %1, off"
               :: "v"(lds_byte_addr), "v"((unsigned long long)(size_t)gaddr)
               : "memory");
}
DEV void async_copy_b128_off16(unsigned lds_byte_addr, const void* gaddr) {
  asm volatile("global_load_async_to_lds_b128 %0, %1, off offset:16"
               :: "v"(lds_byte_addr), "v"((unsigned long long)(size_t)gaddr)
               : "memory");
}
DEV void wait_asynccnt0() {
  asm volatile("s_wait_asynccnt 0x0" ::: "memory");
}
DEV void sched_fence() { asm volatile("" ::: "memory"); }  // compiler-only fence

// ---------------------------------------------------------------- fp32 -> bf16
__global__ void cvt_f32_bf16_kernel(const float* __restrict__ in,
                                    unsigned short* __restrict__ out, int n) {
  int i = blockIdx.x * 256 + threadIdx.x;
  if (i < n) out[i] = f2bf(in[i]);
}

// ---------------------------------------------------------------- bf16 GEMM
// D = scale * (A @ B) + bias [+ resid], A row-major bf16 (optionally split in K
// across two tensors), B row-major bf16 [K][N]. Output fp32 or bf16.
// Block tile 128x128, BK=32, 256 threads = 8 waves (4 along M x 2 along N),
// each wave computes 32x64 via 8 v_wmma_f32_16x16x32_bf16 accumulators.
// Double-buffered LDS pipeline, one barrier per k-chunk: A tile prefetched by
// GLOBAL_LOAD_ASYNC_TO_LDS_B128 (ASYNCcnt) into the idle buffer while the
// current chunk's WMMAs run; next B tile is pulled into VGPRs concurrently.
__global__ __launch_bounds__(256)
void gemm_bf16_kernel(const unsigned short* __restrict__ A0, int lda0, int ksplit,
                      const unsigned short* __restrict__ A1, int lda1,
                      const unsigned short* __restrict__ Bm, int ldb,
                      const float* __restrict__ bias, float scale,
                      const float* __restrict__ resid,
                      float* __restrict__ outF, unsigned short* __restrict__ outH,
                      int M, int N, int K)
{
  __shared__ unsigned short Asm[2][128 * 32];   // row-major [row][k]  (2x8 KB)
  __shared__ unsigned int   Bsm[2][128 * 16];   // pair-packed [n][k2] (2x8 KB)

  const int tid = threadIdx.x;
  const int w = tid >> 5, lane = tid & 31, hl = lane >> 4, lr = lane & 15;
  const int wm = w & 3, wn = w >> 2;
  const int blockRow = blockIdx.y * 128;
  const int blockCol = blockIdx.x * 128;

  v8f acc[2][4] = {};

  const int a_r  = tid >> 1;
  const int a_c  = (tid & 1) * 16;
  const int b_k2 = tid >> 4;
  const int b_n0 = (tid & 15) * 8;

  auto issueA = [&](int kc, int buf) {  // async A stage (32-aligned split)
    int kg = kc + a_c;
    const unsigned short* src =
        (kg < ksplit) ? A0 + (size_t)(blockRow + a_r) * lda0 + kg
                      : A1 + (size_t)(blockRow + a_r) * lda1 + (kg - ksplit);
    unsigned lds = (unsigned)(size_t)&Asm[buf][a_r * 32 + a_c];
    async_copy_b128(lds, src);          // 16 bf16 = 2 x b128
    async_copy_b128_off16(lds, src);
  };
  auto loadB = [&](int kc, HalfView& r0, HalfView& r1) {
    const unsigned short* src = Bm + (size_t)(kc + b_k2 * 2) * ldb + blockCol + b_n0;
    r0.q = *(const u32x4*)src;
    r1.q = *(const u32x4*)(src + ldb);
  };

  HalfView br0, br1;
  issueA(0, 0);
  loadB(0, br0, br1);

  const int T = K >> 5;
  for (int it = 0; it < T; ++it) {
    const int cur = it & 1;
    // commit prefetched B registers pair-packed: Bsm[n][k2]=(B[2k2][n],B[2k2+1][n])
#pragma unroll
    for (int i = 0; i < 8; ++i)
      Bsm[cur][(b_n0 + i) * 16 + b_k2] =
          (unsigned)br0.h[i] | ((unsigned)br1.h[i] << 16);
    wait_asynccnt0();                 // A[cur] DMA landed (per-wave counter)
    __syncthreads();                  // single sync point per k-chunk
    if (it + 1 < T) {                 // overlap next-chunk DMA + global B loads
      issueA((it + 1) << 5, cur ^ 1);
      loadB((it + 1) << 5, br0, br1);
    }

    FragBF16 af[2];
#pragma unroll
    for (int mt = 0; mt < 2; ++mt) {  // A frag: lane = M row, half-wave = K+8
      const unsigned short* p = &Asm[cur][(wm * 32 + mt * 16 + lr) * 32 + 8 * hl];
      af[mt].q[0] = *(const u32x4*)p;
      af[mt].q[1] = *(const u32x4*)(p + 16);
    }
    FragBF16 bfr[4];
#pragma unroll
    for (int nt = 0; nt < 4; ++nt) {  // B frag: packed pairs k2=4h..4h+3, 8+4h..8+4h+3
      const unsigned* p = &Bsm[cur][(wn * 64 + nt * 16 + lr) * 16 + 4 * hl];
      bfr[nt].q[0] = *(const u32x4*)p;
      bfr[nt].q[1] = *(const u32x4*)(p + 8);
    }
#pragma unroll
    for (int mt = 0; mt < 2; ++mt)
#pragma unroll
      for (int nt = 0; nt < 4; ++nt)
        acc[mt][nt] = wmma_bf16(af[mt].v, bfr[nt].v, acc[mt][nt]);
  }

#pragma unroll
  for (int mt = 0; mt < 2; ++mt)
#pragma unroll
    for (int nt = 0; nt < 4; ++nt)
#pragma unroll
      for (int r = 0; r < 8; ++r) {   // C layout: vgpr r -> row r (+8 for hi half)
        int row = blockRow + wm * 32 + mt * 16 + r + 8 * hl;
        int col = blockCol + wn * 64 + nt * 16 + lr;
        float v = acc[mt][nt][r] * scale + bias[col];
        if (resid) v += resid[(size_t)row * N + col];
        if (outH) outH[(size_t)row * N + col] = f2bf(v);
        else      outF[(size_t)row * N + col] = v;
      }
}

// ---------------------------------------------------------------- fused attention
// For every (b,h): row-softmax(Q K^T) @ V with online (flash) normalization.
// Run twice with swapped operands to get both m0 and m1 (column softmax ==
// row softmax of sim^T). qk inputs are pre-scaled by DH^-0.25 each.
// Double-buffered: the 128x64 K slab (128 rows x 128B, stride 1024B) for block
// jb+1 is fetched by the Tensor Data Mover (wave 0 issues tensor_load_to_lds,
// TENSORcnt) while block jb's 32 WMMAs run; V slab prefetch rides in VGPRs.
#define FN  1024
#define FD  512
#define FDH 64

__global__ __launch_bounds__(256)
void flash_cross_kernel(const unsigned short* __restrict__ Qm,
                        const unsigned short* __restrict__ Km,
                        const unsigned short* __restrict__ Vm,
                        unsigned short* __restrict__ Om)
{
  __shared__ unsigned short Ksm[2][128 * 64];   // other-side qk slab (2x16 KB)
  __shared__ unsigned int   Vsm[2][64 * 64];    // V pair-packed over j (2x16 KB)
  __shared__ unsigned short Psm[8 * 16 * 128];  // per-wave P, row-major (32 KB)

  const int tid = threadIdx.x;
  const int w = tid >> 5, lane = tid & 31, hl = lane >> 4, lr = lane & 15;
  const int ib = blockIdx.x;                   // i-block 0..7
  const int bh = blockIdx.y;                   // 0..63
  const int b = bh >> 3, hh = bh & 7;

  const size_t baseQ = (size_t)(b * FN) * FD + hh * FDH;
  const int i0 = ib * 128 + w * 16;            // this wave's 16 rows

  // ---- TDM descriptor group1 (invariant): data_size=4B, tensor 32x128 dwords,
  // tile 32x128, dim0 stride 256 dwords (1024B rows). Bit layout per ISA 8.4.
  unsigned long long q0 = (2ull << 16) | (32ull << 48);          // data_size | tdim0.lo
  unsigned long long q1 = (128ull << 16) | (32ull << 48);        // tdim1 | tile0
  unsigned long long q2 = 128ull | (256ull << 32);               // tile1 | stride0.lo
  unsigned long long q3 = 0;                                     // stride0.hi | stride1
  i32x8 kg1;
  kg1[0] = (int)q0; kg1[1] = (int)(q0 >> 32);
  kg1[2] = (int)q1; kg1[3] = (int)(q1 >> 32);
  kg1[4] = (int)q2; kg1[5] = (int)(q2 >> 32);
  kg1[6] = (int)q3; kg1[7] = (int)(q3 >> 32);
  const i32x4 zero4 = {0, 0, 0, 0};

  auto issueK = [&](int j0, int buf) {  // wave-uniform: call under (w == 0)
    const unsigned long long kga =
        (unsigned long long)(size_t)(Km + baseQ + (size_t)j0 * FD);
    u32x4 g0;
    g0[0] = 1u;                                   // count=1, user mode
    g0[1] = (unsigned)(size_t)(void*)&Ksm[buf][0];// lds_addr
    g0[2] = (unsigned)kga;                        // global_addr[31:0]
    g0[3] = (unsigned)(kga >> 32) | (2u << 30);   // global_addr[56:32] | type=2
    asm volatile("tensor_load_to_lds %0, %1, %2, %3"
                 :: "s"(g0), "s"(kg1), "s"(zero4), "s"(zero4)
                 : "memory");
  };

  const int vk2 = tid >> 2;            // V prefetch: this thread's pair row
  const int vd0 = (tid & 3) * 16;      //   and 16-column chunk (2 tasks of 8)
  auto loadV = [&](int j0, HalfView* vr) {
#pragma unroll
    for (int t = 0; t < 2; ++t) {
      const unsigned short* src =
          Vm + baseQ + (size_t)(j0 + 2 * vk2) * FD + vd0 + 8 * t;
      vr[2 * t + 0].q = *(const u32x4*)src;
      vr[2 * t + 1].q = *(const u32x4*)(src + FD);
    }
  };

  FragBF16 qf[2];                              // Q kept in registers all kernel
  {
    const unsigned short* qrow = Qm + baseQ + (size_t)(i0 + lr) * FD;
#pragma unroll
    for (int s = 0; s < 2; ++s) {
      qf[s].q[0] = *(const u32x4*)(qrow + 32 * s + 8 * hl);
      qf[s].q[1] = *(const u32x4*)(qrow + 32 * s + 16 + 8 * hl);
    }
  }

  v8f o[4] = {};
  float mrow[8], lrow[8];
#pragma unroll
  for (int r = 0; r < 8; ++r) { mrow[r] = -1e30f; lrow[r] = 0.f; }

  HalfView vr[4];
  if (w == 0) issueK(0, 0);
  loadV(0, vr);

  for (int jb = 0; jb < 8; ++jb) {
    const int cur = jb & 1;
    // commit prefetched V registers pair-packed over j: Vsm[d][j2]
#pragma unroll
    for (int t = 0; t < 2; ++t)
#pragma unroll
      for (int i = 0; i < 8; ++i)
        Vsm[cur][(vd0 + 8 * t + i) * 64 + vk2] =
            (unsigned)vr[2 * t + 0].h[i] | ((unsigned)vr[2 * t + 1].h[i] << 16);
    if (w == 0) __builtin_amdgcn_s_wait_tensorcnt(0);  // K[cur] DMA done
    __syncthreads();                   // single sync point per j-block
    if (jb + 1 < 8) {                  // overlap next slab's TDM + V loads
      if (w == 0) issueK((jb + 1) * 128, cur ^ 1);
      loadV((jb + 1) * 128, vr);
    }

    // S = Q @ K^T : 16x128 per wave, 8 col-tiles x 2 K-steps
    v8f S[8];
    const unsigned* Ku = (const unsigned*)&Ksm[cur][0];
#pragma unroll
    for (int nt = 0; nt < 8; ++nt) {
      v8f c = {};
#pragma unroll
      for (int s = 0; s < 2; ++s) {
        FragBF16 kb;
        const unsigned* p = &Ku[(nt * 16 + lr) * 32 + 16 * s + 4 * hl];
        kb.q[0] = *(const u32x4*)p;
        kb.q[1] = *(const u32x4*)(p + 8);
        c = wmma_bf16(qf[s].v, kb.v, c);
      }
      S[nt] = c;
    }

    // online softmax per row (rows live in the 16-lane half per C layout)
#pragma unroll
    for (int r = 0; r < 8; ++r) {
      float mx = S[0][r];
#pragma unroll
      for (int nt = 1; nt < 8; ++nt) mx = fmaxf(mx, S[nt][r]);
#pragma unroll
      for (int off = 8; off >= 1; off >>= 1)
        mx = fmaxf(mx, __shfl_xor(mx, off, 32));
      float mn = fmaxf(mrow[r], mx);
      float fct = __expf(mrow[r] - mn);
      mrow[r] = mn;
      lrow[r] *= fct;
#pragma unroll
      for (int nt = 0; nt < 4; ++nt) o[nt][r] *= fct;
      float sum = 0.f;
#pragma unroll
      for (int nt = 0; nt < 8; ++nt) {
        float pv = __expf(S[nt][r] - mn);
        sum += pv;
        Psm[w * 2048 + (r + 8 * hl) * 128 + nt * 16 + lr] = f2bf(pv);
      }
#pragma unroll
      for (int off = 8; off >= 1; off >>= 1) sum += __shfl_xor(sum, off, 32);
      lrow[r] += sum;
    }
    sched_fence();   // P is wave-private; same-wave DS ops are in-order, just
                     // stop the compiler from hoisting the reads below.

    // O += P @ V : K=128 -> 4 K-steps x 4 col-tiles
#pragma unroll
    for (int s = 0; s < 4; ++s) {
      FragBF16 pa;
      const unsigned short* pp = &Psm[w * 2048 + lr * 128 + 32 * s + 8 * hl];
      pa.q[0] = *(const u32x4*)pp;
      pa.q[1] = *(const u32x4*)(pp + 16);
#pragma unroll
      for (int nt = 0; nt < 4; ++nt) {
        FragBF16 vb;
        const unsigned* p = &Vsm[cur][(nt * 16 + lr) * 64 + 16 * s + 4 * hl];
        vb.q[0] = *(const u32x4*)p;
        vb.q[1] = *(const u32x4*)(p + 8);
        o[nt] = wmma_bf16(pa.v, vb.v, o[nt]);
      }
    }
  }

#pragma unroll
  for (int r = 0; r < 8; ++r) {
    float inv = 1.0f / lrow[r];
    int row = i0 + r + 8 * hl;
#pragma unroll
    for (int nt = 0; nt < 4; ++nt)
      Om[baseQ + (size_t)row * FD + nt * 16 + lr] = f2bf(o[nt][r] * inv);
  }
}

// ---------------------------------------------------------------- LayerNorm + GELU
__global__ __launch_bounds__(256)
void ln_gelu_kernel(const float* __restrict__ Y, const float* __restrict__ g,
                    const float* __restrict__ bta, unsigned short* __restrict__ outH)
{
  __shared__ float red[16];
  const int row = blockIdx.x;
  const int tid = threadIdx.x;
  const float* yr = Y + (size_t)row * 1024;
  float yv[4], s = 0.f, ss = 0.f;
#pragma unroll
  for (int i = 0; i < 4; ++i) {
    yv[i] = yr[tid + 256 * i];
    s += yv[i]; ss += yv[i] * yv[i];
  }
#pragma unroll
  for (int off = 16; off >= 1; off >>= 1) {
    s  += __shfl_xor(s,  off, 32);
    ss += __shfl_xor(ss, off, 32);
  }
  if ((tid & 31) == 0) { red[tid >> 5] = s; red[8 + (tid >> 5)] = ss; }
  __syncthreads();
  s = 0.f; ss = 0.f;
#pragma unroll
  for (int i = 0; i < 8; ++i) { s += red[i]; ss += red[8 + i]; }
  const float mu   = s * (1.f / 1024.f);
  const float var  = ss * (1.f / 1024.f) - mu * mu;
  const float rstd = rsqrtf(var + 1e-5f);
  unsigned short* orow = outH + (size_t)row * 1024;
#pragma unroll
  for (int i = 0; i < 4; ++i) {
    int c = tid + 256 * i;
    float v = (yv[i] - mu) * rstd * g[c] + bta[c];
    v = 0.5f * v * (1.f + erff(v * 0.70710678118654752f));
    orow[c] = f2bf(v);
  }
}

// ---------------------------------------------------------------- host driver
extern "C" void kernel_launch(void* const* d_in, const int* in_sizes, int n_in,
                              void* d_out, int out_size, void* d_ws, size_t ws_size,
                              hipStream_t stream)
{
  (void)in_sizes; (void)n_in; (void)out_size; (void)ws_size;
  const float* x0  = (const float*)d_in[0];
  const float* x1  = (const float*)d_in[1];
  const float* Wqk = (const float*)d_in[2];
  const float* bqk = (const float*)d_in[3];
  const float* Wv  = (const float*)d_in[4];
  const float* bv  = (const float*)d_in[5];
  const float* Wo  = (const float*)d_in[6];
  const float* bo  = (const float*)d_in[7];
  const float* Wf1 = (const float*)d_in[8];
  const float* bf1 = (const float*)d_in[9];
  const float* lng = (const float*)d_in[10];
  const float* lnb = (const float*)d_in[11];
  const float* Wf2 = (const float*)d_in[12];
  const float* bf2 = (const float*)d_in[13];
  float* out = (float*)d_out;

  const int BN = 8 * 1024;  // 8192 token rows
  char* p = (char*)d_ws;
  auto carve = [&](size_t bytes) -> unsigned short* {
    char* r = p; p += (bytes + 255) & ~(size_t)255; return (unsigned short*)r;
  };
  unsigned short* xb0  = carve((size_t)BN * 512 * 2);
  unsigned short* xb1  = carve((size_t)BN * 512 * 2);
  unsigned short* wqkb = carve((size_t)512 * 512 * 2);
  unsigned short* wvb  = carve((size_t)512 * 512 * 2);
  unsigned short* wob  = carve((size_t)512 * 512 * 2);
  unsigned short* wf1b = carve((size_t)1024 * 1024 * 2);
  unsigned short* wf2b = carve((size_t)1024 * 512 * 2);
  unsigned short* qk0b = carve((size_t)BN * 512 * 2);
  unsigned short* qk1b = carve((size_t)BN * 512 * 2);
  unsigned short* v0b  = carve((size_t)BN * 512 * 2);
  unsigned short* v1b  = carve((size_t)BN * 512 * 2);
  unsigned short* am0  = carve((size_t)BN * 512 * 2);
  unsigned short* am1  = carve((size_t)BN * 512 * 2);
  unsigned short* m0p  = carve((size_t)BN * 512 * 2);
  unsigned short* m1p  = carve((size_t)BN * 512 * 2);
  float*          yb   = (float*)carve((size_t)BN * 1024 * 4);
  unsigned short* gb   = carve((size_t)BN * 1024 * 2);

  auto cvt = [&](const float* src, unsigned short* dst, int n) {
    cvt_f32_bf16_kernel<<<(n + 255) / 256, 256, 0, stream>>>(src, dst, n);
  };
  auto gemm = [&](const unsigned short* A0, int lda0, int ksplit,
                  const unsigned short* A1, int lda1,
                  const unsigned short* B, int ldb, const float* bias,
                  float scale, const float* resid,
                  float* oF, unsigned short* oH, int M, int N, int K) {
    dim3 grid(N / 128, M / 128);
    gemm_bf16_kernel<<<grid, 256, 0, stream>>>(A0, lda0, ksplit, A1, lda1, B, ldb,
                                               bias, scale, resid, oF, oH, M, N, K);
  };

  cvt(x0, xb0, BN * 512);
  cvt(x1, xb1, BN * 512);
  cvt(Wqk, wqkb, 512 * 512);
  cvt(Wv,  wvb,  512 * 512);
  cvt(Wo,  wob,  512 * 512);
  cvt(Wf1, wf1b, 1024 * 1024);
  cvt(Wf2, wf2b, 1024 * 512);

  const float SC = 0.35355339059327379f;  // DH^-0.25, folded into projection
  gemm(xb0, 512, 512, xb0, 512, wqkb, 512, bqk, SC,  nullptr, nullptr, qk0b, BN, 512, 512);
  gemm(xb1, 512, 512, xb1, 512, wqkb, 512, bqk, SC,  nullptr, nullptr, qk1b, BN, 512, 512);
  gemm(xb0, 512, 512, xb0, 512, wvb,  512, bv,  1.f, nullptr, nullptr, v0b,  BN, 512, 512);
  gemm(xb1, 512, 512, xb1, 512, wvb,  512, bv,  1.f, nullptr, nullptr, v1b,  BN, 512, 512);

  flash_cross_kernel<<<dim3(8, 64), 256, 0, stream>>>(qk0b, qk1b, v1b, am0);
  flash_cross_kernel<<<dim3(8, 64), 256, 0, stream>>>(qk1b, qk0b, v0b, am1);

  gemm(am0, 512, 512, am0, 512, wob, 512, bo, 1.f, nullptr, nullptr, m0p, BN, 512, 512);
  gemm(am1, 512, 512, am1, 512, wob, 512, bo, 1.f, nullptr, nullptr, m1p, BN, 512, 512);

  // branch 0: FFN(concat(x0, m0)) + residual
  gemm(xb0, 512, 512, m0p, 512, wf1b, 1024, bf1, 1.f, nullptr, yb, nullptr, BN, 1024, 1024);
  ln_gelu_kernel<<<BN, 256, 0, stream>>>(yb, lng, lnb, gb);
  gemm(gb, 1024, 1024, gb, 1024, wf2b, 512, bf2, 1.f, x0, out, nullptr, BN, 512, 1024);

  // branch 1: FFN(concat(x1, m1)) + residual
  gemm(xb1, 512, 512, m1p, 512, wf1b, 1024, bf1, 1.f, nullptr, yb, nullptr, BN, 1024, 1024);
  ln_gelu_kernel<<<BN, 256, 0, stream>>>(yb, lng, lnb, gb);
  gemm(gb, 1024, 1024, gb, 1024, wf2b, 512, bf2, 1.f, x1,
       out + (size_t)BN * 512, nullptr, BN, 512, 1024);
}
